// GM_SV_V2_22119081575277
// MI455X (gfx1250) — compile-verified, matrix operand order checked by reading, not run on hardware
//
#include <hip/hip_runtime.h>
#include <hip/hip_bf16.h>

typedef __attribute__((ext_vector_type(16))) _Float16 v16h;
typedef __attribute__((ext_vector_type(8)))  float    v8f;

#define B_ 16
#define NL 48
#define NT 512
#define H_ 128
#define G_ 10
#define NATOMS 15

// output layout (floats), concatenated in reference return order
#define OFF_PI   0
#define OFF_SIG  (B_*NL*NT*G_)
#define OFF_MU   (2*B_*NL*NT*G_)
#define OFF_DIST (3*B_*NL*NT*G_)
#define OFF_MASK (3*B_*NL*NT*G_ + B_*NL*NT)

// workspace layout (bytes)
#define WS_AL    0
#define WS_AT    (WS_AL + B_*NL*H_*4)
#define WS_BIAS  (WS_AT + B_*NT*H_*4)
#define WS_BCAT  (WS_BIAS + H_*4)
#define WS_WFRAG (WS_BCAT + 32*4)

// CDNA5 async-copy completion wait (ASYNCcnt)
#if defined(__has_builtin)
#if __has_builtin(__builtin_amdgcn_s_wait_asynccnt)
#define WAIT_ASYNC0() __builtin_amdgcn_s_wait_asynccnt(0)
#else
#define WAIT_ASYNC0() asm volatile("s_wait_asynccnt 0" ::: "memory")
#endif
#else
#define WAIT_ASYNC0() asm volatile("s_wait_asynccnt 0" ::: "memory")
#endif

// ---------------------------------------------------------------------------
// Prep 1: fold BN into bias, build concatenated head bias, and pre-swizzle the
// concatenated head weights W_cat[128x32] into the exact per-lane v16h B-matrix
// fragment layout of V_WMMA_F32_16X16X32_F16 (wave32).
// ---------------------------------------------------------------------------
__global__ void prep_fold(const float* __restrict__ b1,
                          const float* __restrict__ gamma,
                          const float* __restrict__ beta,
                          const float* __restrict__ mean,
                          const float* __restrict__ var,
                          const float* __restrict__ Wpi,  const float* __restrict__ bpi,
                          const float* __restrict__ Wsig, const float* __restrict__ bsig,
                          const float* __restrict__ Wmu,  const float* __restrict__ bmu,
                          float* __restrict__ bias, float* __restrict__ bcat,
                          _Float16* __restrict__ wfrag) {
  int t = threadIdx.x;
  if (t < H_) {
    float s = rsqrtf(var[t] + 1e-5f) * gamma[t];
    bias[t] = (b1[t] - mean[t]) * s + beta[t];
  }
  if (t < 32) {
    float v = 0.f;
    if (t < 10)      v = bpi[t];
    else if (t < 20) v = bsig[t - 10];
    else if (t < 30) v = bmu[t - 20];
    bcat[t] = v;
  }
  // 256 threads -> (ntile[2], ktile[4], lane[32]); each writes 16 halves
  int lane  = t & 31;
  int ktile = (t >> 5) & 3;
  int ntile = t >> 7;
  int n = ntile * 16 + (lane & 15);
  int kbase = ktile * 32 + ((lane & 16) ? 8 : 0);
  for (int e = 0; e < 16; ++e) {
    int k = kbase + (e < 8 ? e : e + 8);
    float v = 0.f;
    if (n < 10)      v = Wpi[k * G_ + n];
    else if (n < 20) v = Wsig[k * G_ + (n - 10)];
    else if (n < 30) v = Wmu[k * G_ + (n - 20)];
    wfrag[((ntile * 4 + ktile) * 32 + lane) * 16 + e] = (_Float16)v;
  }
}

// ---------------------------------------------------------------------------
// Prep 2: A_l = (h_l_x @ W1) * s   and   A_t = (h_t_x @ W1) * s
// one block per output row; thread k = output feature; W1 reads coalesced.
// ---------------------------------------------------------------------------
__global__ __launch_bounds__(128)
void prep_axw(const float* __restrict__ hlx, const float* __restrict__ htx,
              const float* __restrict__ W1,
              const float* __restrict__ gamma, const float* __restrict__ var,
              float* __restrict__ Al, float* __restrict__ At) {
  __shared__ float xs[H_];
  int row = blockIdx.x;
  int k = threadIdx.x;
  const float* x = (row < B_ * NL) ? (hlx + (long)row * H_)
                                   : (htx + (long)(row - B_ * NL) * H_);
  xs[k] = x[k];
  __syncthreads();
  float acc = 0.f;
#pragma unroll 8
  for (int h = 0; h < H_; ++h) acc += xs[h] * W1[h * H_ + k];
  float s = rsqrtf(var[k] + 1e-5f) * gamma[k];
  float* out = (row < B_ * NL) ? (Al + (long)row * H_)
                               : (At + (long)(row - B_ * NL) * H_);
  out[k] = acc * s;
}

// ---------------------------------------------------------------------------
// Main fused kernel: per (b,i) and 64-wide j chunk.
// 4 waves; each wave: 16 pairs x 32 logits via 8x v_wmma_f32_16x16x32_f16.
// A_t tile staged via CDNA5 async global->LDS copies (no VGPR round trip).
// Epilogue: softmax(pi) / elu(sigma)+1.1 / elu(mu)+1.0, masked, write out.
// ---------------------------------------------------------------------------
__global__ __launch_bounds__(128)
void pair_mdn(const float* __restrict__ Al, const float* __restrict__ At,
              const float* __restrict__ bias, const float* __restrict__ bcat,
              const _Float16* __restrict__ wfrag,
              const unsigned char* __restrict__ lmask,
              const unsigned char* __restrict__ tmask,
              float* __restrict__ out) {
  __shared__ float lat[64 * H_];       // A_t tile (32 KB)
  __shared__ float lal[H_];
  __shared__ float lbias[H_];
  __shared__ float lbc[32];
  __shared__ float llog[4 * 16 * 32];  // per-wave logits staging

  int jchunk = blockIdx.x & 7;
  int bi = blockIdx.x >> 3;
  int b = bi / NL, i = bi % NL;
  int jbase = jchunk * 64;
  int tid = threadIdx.x;

  // stage A_t tile: async global->LDS b128 copies, tracked by ASYNCcnt.
  // GVS addressing: uniform SGPR base + per-lane byte offset; LDS destination
  // address comes from the low 32 bits of the generic shared-memory pointer.
  {
    const char* gbase = (const char*)(At + (long)(b * NT + jbase) * H_);
    unsigned ldsbase = (unsigned)(unsigned long long)(&lat[0]);
#pragma unroll
    for (int it = 0; it < 16; ++it) {
      unsigned off = (unsigned)((it * 128 + tid) * 16);
      unsigned laddr = ldsbase + off;
      asm volatile("global_load_async_to_lds_b128 %0, %1, %2"
                   :
                   : "v"(laddr), "v"(off), "s"(gbase)
                   : "memory");
    }
  }
  lal[tid]   = Al[(long)(b * NL + i) * H_ + tid];
  lbias[tid] = bias[tid];
  if (tid < 32) lbc[tid] = bcat[tid];
  WAIT_ASYNC0();
  __syncthreads();

  int wave = tid >> 5, lane = tid & 31;
  int row = lane & 15, hlf = lane >> 4;
  int rbase = wave * 16;

  // Build f16 A fragments: h = elu(A_l + A_t + bias), ISA 16-bit A layout
  v16h af[4];
  const float* atrow = lat + (rbase + row) * H_;
#pragma unroll
  for (int kt = 0; kt < 4; ++kt) {
    int kb = kt * 32 + hlf * 8;
#pragma unroll
    for (int e = 0; e < 16; ++e) {
      int k = kb + (e < 8 ? e : e + 8);
      float v = atrow[k] + lal[k] + lbias[k];
      v = v > 0.f ? v : (__expf(v) - 1.0f);
      af[kt][e] = (_Float16)v;
    }
  }

  // 8 WMMAs: D = A x W_cat (+acc)
  const v16h* wf = (const v16h*)wfrag;
  v8f acc0 = {}, acc1 = {};
#pragma unroll
  for (int kt = 0; kt < 4; ++kt) {
    v16h w0 = wf[(0 * 4 + kt) * 32 + lane];
    v16h w1 = wf[(1 * 4 + kt) * 32 + lane];
    acc0 = __builtin_amdgcn_wmma_f32_16x16x32_f16(false, af[kt], false, w0,
                                                  (short)0, acc0, false, false);
    acc1 = __builtin_amdgcn_wmma_f32_16x16x32_f16(false, af[kt], false, w1,
                                                  (short)0, acc1, false, false);
  }

  // spill C layout (VGPR r: M=r+8*hlf, N=lane%16) to LDS for row-major epilogue
  float* lg = llog + wave * 512;
#pragma unroll
  for (int r = 0; r < 8; ++r) {
    lg[(r + hlf * 8) * 32 + row]      = acc0[r];
    lg[(r + hlf * 8) * 32 + 16 + row] = acc1[r];
  }
  __syncthreads();

  if (hlf == 0) {  // lanes 0..15: one pair-row each
    int j = jbase + rbase + row;
    long p = ((long)(b * NL + i)) * NT + j;
    float v[32];
#pragma unroll
    for (int n = 0; n < 32; ++n) v[n] = lg[row * 32 + n] + lbc[n];
    float m = (lmask[b * NL + i] && tmask[b * NT + j]) ? 1.0f : 0.0f;

    float mx = v[0];
#pragma unroll
    for (int g = 1; g < 10; ++g) mx = fmaxf(mx, v[g]);
    float ex[10], sum = 0.f;
#pragma unroll
    for (int g = 0; g < 10; ++g) { ex[g] = __expf(v[g] - mx); sum += ex[g]; }
    float inv = 1.0f / sum;
#pragma unroll
    for (int g = 0; g < 10; ++g) out[OFF_PI + p * 10 + g] = m * ex[g] * inv;
#pragma unroll
    for (int g = 0; g < 10; ++g) {
      float x = v[10 + g];
      x = x > 0.f ? x : (__expf(x) - 1.0f);
      out[OFF_SIG + p * 10 + g] = m * (x + 1.1f);
    }
#pragma unroll
    for (int g = 0; g < 10; ++g) {
      float x = v[20 + g];
      x = x > 0.f ? x : (__expf(x) - 1.0f);
      out[OFF_MU + p * 10 + g] = m * (x + 1.0f);
    }
  }
}

// ---------------------------------------------------------------------------
// Min res-res distance + mask output. One block per (b,i); thread handles 2 j.
// min over squared distances (monotone under sqrt; matches reference's
// NaN-excluded min since direct-diff d2 >= 0).
// ---------------------------------------------------------------------------
__global__ __launch_bounds__(256)
void dist_kernel(const float* __restrict__ lpos, const float* __restrict__ tpos,
                 const unsigned char* __restrict__ lmask,
                 const unsigned char* __restrict__ tmask,
                 float* __restrict__ out) {
  __shared__ float la[NATOMS * 3];
  int bi = blockIdx.x;
  int b = bi / NL;
  int tid = threadIdx.x;
  if (tid < NATOMS * 3) la[tid] = lpos[(long)bi * NATOMS * 3 + tid];
  __syncthreads();
  unsigned char lm = lmask[bi];
  for (int jj = 0; jj < 2; ++jj) {
    int j = tid + jj * 256;
    const float* tp = tpos + ((long)(b * NT + j)) * NATOMS * 3;
    float ta[NATOMS * 3];
#pragma unroll
    for (int a = 0; a < NATOMS * 3; ++a) ta[a] = tp[a];
    float mind2 = 3.4e38f;
    for (int al = 0; al < NATOMS; ++al) {
      float x = la[al * 3], y = la[al * 3 + 1], z = la[al * 3 + 2];
#pragma unroll
      for (int at = 0; at < NATOMS; ++at) {
        float dx = x - ta[at * 3];
        float dy = y - ta[at * 3 + 1];
        float dz = z - ta[at * 3 + 2];
        float d2 = dx * dx + dy * dy + dz * dz;
        mind2 = fminf(mind2, d2);
      }
    }
    float m = (lm && tmask[b * NT + j]) ? 1.0f : 0.0f;
    long p = (long)bi * NT + j;
    out[OFF_DIST + p] = m * sqrtf(mind2);
    out[OFF_MASK + p] = m;
  }
}

// ---------------------------------------------------------------------------
extern "C" void kernel_launch(void* const* d_in, const int* in_sizes, int n_in,
                              void* d_out, int out_size, void* d_ws, size_t ws_size,
                              hipStream_t stream) {
  const float* hlx  = (const float*)d_in[0];
  const float* htx  = (const float*)d_in[1];
  const unsigned char* lm = (const unsigned char*)d_in[2];
  const unsigned char* tm = (const unsigned char*)d_in[3];
  const float* lpos = (const float*)d_in[4];
  const float* tpos = (const float*)d_in[5];
  const float* W1   = (const float*)d_in[6];
  const float* b1   = (const float*)d_in[7];
  const float* gamma= (const float*)d_in[8];
  const float* beta = (const float*)d_in[9];
  const float* mean = (const float*)d_in[10];
  const float* var  = (const float*)d_in[11];
  const float* Wpi  = (const float*)d_in[12];
  const float* bpi  = (const float*)d_in[13];
  const float* Wsig = (const float*)d_in[14];
  const float* bsig = (const float*)d_in[15];
  const float* Wmu  = (const float*)d_in[16];
  const float* bmu  = (const float*)d_in[17];

  char* ws = (char*)d_ws;
  float* Al   = (float*)(ws + WS_AL);
  float* At   = (float*)(ws + WS_AT);
  float* bias = (float*)(ws + WS_BIAS);
  float* bcat = (float*)(ws + WS_BCAT);
  _Float16* wfrag = (_Float16*)(ws + WS_WFRAG);
  float* out = (float*)d_out;

  prep_fold<<<1, 256, 0, stream>>>(b1, gamma, beta, mean, var,
                                   Wpi, bpi, Wsig, bsig, Wmu, bmu,
                                   bias, bcat, wfrag);
  prep_axw<<<B_ * NL + B_ * NT, 128, 0, stream>>>(hlx, htx, W1, gamma, var, Al, At);
  pair_mdn<<<B_ * NL * 8, 128, 0, stream>>>(Al, At, bias, bcat, wfrag, lm, tm, out);
  dist_kernel<<<B_ * NL, 256, 0, stream>>>(lpos, tpos, lm, tm, out);
}
